// EnhancedSpikingRetrievalCore_49718541418725
// MI455X (gfx1250) — matrix operand order repaired and checked by reading, not compiled, not run on hardware
//
#include <hip/hip_runtime.h>

// ---------------- problem constants ----------------
#define B_  8192
#define D_  2048
#define E_  8
#define G_  4
#define P_  512
#define H_  192
#define KROUTE_ 2
#define PREDW_ 0.1f
#define GAIN_MEAN 0.015625f   // KTOP/D: LIF block collapses to this constant

#define MTILE 32              // rows per workgroup (two 16-row WMMA tiles)
#define MAX_TILES 256         // worst case B/MTILE row-tiles per expert
#define CHUNK 256             // K-chunk of x staged in LDS

// ---------------- WMMA types ----------------
typedef __attribute__((ext_vector_type(16))) __bf16 v16bf;
typedef __attribute__((ext_vector_type(8)))  float  v8f;
typedef __attribute__((ext_vector_type(4)))  unsigned int u32x4;

union BFV { v16bf v; u32x4 q[2]; };

// ---------------- helpers ----------------
__device__ __forceinline__ unsigned short f2bf(float f) {
  union { float f; unsigned u; } v; v.f = f;
  unsigned u = v.u;
  u += 0x7fffu + ((u >> 16) & 1u);         // round-to-nearest-even
  return (unsigned short)(u >> 16);
}
__device__ __forceinline__ unsigned pack2(float a, float b) {
  return (unsigned)f2bf(a) | ((unsigned)f2bf(b) << 16);
}

// A-operand (MxK=16x32 bf16): lane m=L%16, half=L/16;
// VGPR0-7: K = half*16 + 0..15 (contiguous, matches load_b)
__device__ __forceinline__ v16bf load_a(const unsigned short* rowk, int half) {
  BFV t;
  t.q[0] = *(const u32x4*)(rowk + half * 16);
  t.q[1] = *(const u32x4*)(rowk + half * 16 + 8);
  return t.v;
}
// B-operand (KxN=32x16 bf16): lane n=L%16, half=L/16;
// VGPR v holds K = half*16 + 2v, 2v+1  -> 16 contiguous bf16 per lane
__device__ __forceinline__ v16bf load_b(const unsigned short* colk_half) {
  BFV t;
  t.q[0] = *(const u32x4*)(colk_half);
  t.q[1] = *(const u32x4*)(colk_half + 8);
  return t.v;
}

// ---------------- kernel 0: zero out + counts ----------------
__global__ void init_kernel(float* out, int* counts) {
  if (blockIdx.x == 0 && threadIdx.x < E_) counts[threadIdx.x] = 0;
  const size_t n4 = (size_t)B_ * D_ / 4;
  float4 z = make_float4(0.f, 0.f, 0.f, 0.f);
  for (size_t i = (size_t)blockIdx.x * blockDim.x + threadIdx.x; i < n4;
       i += (size_t)gridDim.x * blockDim.x)
    ((float4*)out)[i] = z;
}

// ---------------- kernel 1: gating (one wave per row) ----------------
__global__ void gate_kernel(const float* __restrict__ x,
                            const float* __restrict__ Wg, const float* __restrict__ bg,
                            const float* __restrict__ Wp, const float* __restrict__ bp,
                            const float* __restrict__ Wgg, const float* __restrict__ bgg,
                            int* __restrict__ rowE, float* __restrict__ rowW,
                            int* __restrict__ counts) {
  const int wave = threadIdx.x >> 5;
  const int lane = threadIdx.x & 31;
  const int row  = blockIdx.x * 8 + wave;
  if (row >= B_) return;

  // row mean
  const float* xr = x + (size_t)row * D_;
  float s = 0.f;
  for (int i = lane; i < D_; i += 32) s += xr[i];
  for (int m = 16; m >= 1; m >>= 1) s += __shfl_xor(s, m, 32);
  const float qm = s * (1.0f / D_);

  // phasor-bank mean: (sum cos + sum sin)/(2H)
  float ts = 0.f;
  for (int h = lane + 1; h <= H_; h += 32) {
    float a = qm * 7.0f * (float)h;
    float sv, cv;
    __sincosf(a, &sv, &cv);
    ts += sv + cv;
  }
  for (int m = 16; m >= 1; m >>= 1) ts += __shfl_xor(ts, m, 32);
  const float gi0 = ts * (1.0f / (2.0f * H_));
  const float gi1 = GAIN_MEAN;

  // 8 logits on lanes 0..7
  float l = -1e30f;
  if (lane < E_) {
    const int e = lane, g = lane & (G_ - 1);
    float lg = gi0 * Wg[e]  + gi1 * Wg[E_ + e]  + bg[e];
    float lp = gi0 * Wp[e]  + gi1 * Wp[E_ + e]  + bp[e];
    float lG = gi0 * Wgg[g] + gi1 * Wgg[G_ + g] + bgg[g];
    l = lg - PREDW_ * lp + lG;
  }
  // top-2 (uniform shfl loop)
  float best = -1e30f, second = -1e30f; int be = 0, se = 1;
  for (int j = 0; j < E_; ++j) {
    float lj = __shfl(l, j, 32);
    if (lj > best)        { second = best; se = be; best = lj; be = j; }
    else if (lj > second) { second = lj; se = j; }
  }
  if (lane == 0) {
    float wa = 1.0f / (1.0f + __expf(second - best));  // renormalized softmax over top-2
    rowE[2 * row] = be; rowE[2 * row + 1] = se;
    rowW[2 * row] = wa; rowW[2 * row + 1] = 1.0f - wa;
    atomicAdd(&counts[be], 1);
    atomicAdd(&counts[se], 1);
  }
}

// ---------------- kernel 2: 8-entry scan ----------------
__global__ void scan_kernel(const int* __restrict__ counts, int* __restrict__ offsets,
                            int* __restrict__ cursor) {
  if (threadIdx.x == 0) {
    int s = 0;
    for (int e = 0; e < E_; ++e) { offsets[e] = s; cursor[e] = s; s += counts[e]; }
    offsets[E_] = s;
  }
}

// ---------------- kernel 3: scatter to expert-major ----------------
__global__ void scatter_kernel(const int* __restrict__ rowE, const float* __restrict__ rowW,
                               int* __restrict__ cursor,
                               int* __restrict__ perm, float* __restrict__ permW) {
  int r = blockIdx.x * blockDim.x + threadIdx.x;
  if (r >= B_) return;
  for (int s2 = 0; s2 < KROUTE_; ++s2) {
    int e = rowE[2 * r + s2];
    int pos = atomicAdd(&cursor[e], 1);
    perm[pos]  = r;
    permW[pos] = rowW[2 * r + s2];
  }
}

// ---------------- kernel 4: fp32 [R,C] -> bf16 [C,R] per expert ----------------
__global__ void transpose_bf16(const float* __restrict__ src, unsigned short* __restrict__ dst,
                               int R, int C) {
  __shared__ float tile[32][33];
  const int e = blockIdx.z;
  const float* s = src + (size_t)e * R * C;
  unsigned short* d = dst + (size_t)e * R * C;
  const int c0 = blockIdx.x * 32, r0 = blockIdx.y * 32;
  for (int i = threadIdx.y; i < 32; i += 8)
    tile[i][threadIdx.x] = s[(size_t)(r0 + i) * C + c0 + threadIdx.x];
  __syncthreads();
  for (int i = threadIdx.y; i < 32; i += 8)
    d[(size_t)(c0 + i) * R + r0 + threadIdx.x] = f2bf(tile[threadIdx.x][i]);
}

// ---------------- kernel 5: routed MoE MLP via bf16 WMMA, M-tile = 32 ----------------
__launch_bounds__(256)
__global__ void moe_gemm_kernel(const float* __restrict__ x,
                                const unsigned short* __restrict__ W1t,  // [E,P,D] bf16
                                const float* __restrict__ b1,            // [E,P]
                                const unsigned short* __restrict__ W2t,  // [E,D,P] bf16
                                const float* __restrict__ b2,            // [E,D]
                                const int* __restrict__ counts, const int* __restrict__ offsets,
                                const int* __restrict__ perm, const float* __restrict__ permW,
                                float* __restrict__ out) {
  __shared__ __align__(16) unsigned short xs[MTILE * CHUNK];  // 32 x 256 bf16 = 16 KB
  __shared__ __align__(16) unsigned short hs[MTILE * P_];     // 32 x 512 bf16 = 32 KB
  __shared__ int   rowids[MTILE];
  __shared__ float wts[MTILE];

  const int e    = blockIdx.x >> 8;             // / MAX_TILES
  const int tile = blockIdx.x & (MAX_TILES - 1);
  const int cnt  = counts[e];
  const int m0   = tile * MTILE;
  if (m0 >= cnt) return;
  const int off = offsets[e];

  const int tid = threadIdx.x;
  if (tid < MTILE) {
    int m = m0 + tid;
    if (m < cnt) { rowids[tid] = perm[off + m]; wts[tid] = permW[off + m]; }
    else         { rowids[tid] = -1;            wts[tid] = 0.0f; }
  }
  __syncthreads();

  const int wave = tid >> 5, lane = tid & 31;
  const int half = lane >> 4, ln = lane & 15;

  // ---------- GEMM1: h[32,P] = relu(x[32,D] @ W1[D,P] + b1) * w ----------
  // wave w covers P columns [w*64, w*64+64) for BOTH 16-row m-tiles (B reuse x2)
  v8f acc[2][4];
#pragma unroll
  for (int mt = 0; mt < 2; ++mt)
    for (int nt = 0; nt < 4; ++nt)
      for (int r = 0; r < 8; ++r) acc[mt][nt][r] = 0.f;

  const unsigned short* W1e = W1t + (size_t)e * P_ * D_;
  const int srow = tid >> 3;              // staging: row 0..31
  const int scol = (tid & 7) * 32;        // 32 elems per thread per chunk

  for (int kc = 0; kc < D_; kc += CHUNK) {
    // stage x chunk -> bf16 LDS (gathered rows)
    {
      int rg = rowids[srow];
      unsigned short* dstp = &xs[srow * CHUNK + scol];
      if (rg >= 0) {
        const float* xr = x + (size_t)rg * D_ + kc + scol;
        if (kc + CHUNK < D_) __builtin_prefetch(xr + CHUNK, 0, 1);
        for (int i = 0; i < 32; i += 4) {
          float4 v = *(const float4*)(xr + i);
          uint2 p; p.x = pack2(v.x, v.y); p.y = pack2(v.z, v.w);
          *(uint2*)(dstp + i) = p;
        }
      } else {
        uint2 z; z.x = 0u; z.y = 0u;
        for (int i = 0; i < 32; i += 4) *(uint2*)(dstp + i) = z;
      }
    }
    __syncthreads();

    const unsigned short* xrow0 = &xs[ln * CHUNK];            // m-tile 0
    const unsigned short* xrow1 = &xs[(16 + ln) * CHUNK];     // m-tile 1
    for (int kk = 0; kk < CHUNK; kk += 32) {
      v16bf a0 = load_a(xrow0 + kk, half);
      v16bf a1 = load_a(xrow1 + kk, half);
#pragma unroll
      for (int nt = 0; nt < 4; ++nt) {
        int ncol = wave * 64 + nt * 16 + ln;
        v16bf b = load_b(W1e + (size_t)ncol * D_ + kc + kk + half * 16);
        acc[0][nt] = __builtin_amdgcn_wmma_f32_16x16x32_bf16(
            false, a0, false, b, (short)0, acc[0][nt], false, false);
        acc[1][nt] = __builtin_amdgcn_wmma_f32_16x16x32_bf16(
            false, a1, false, b, (short)0, acc[1][nt], false, false);
      }
    }
    __syncthreads();   // xs reused next chunk
  }

  // epilogue: relu + bias, scale by gate weight, store bf16 h
#pragma unroll
  for (int mt = 0; mt < 2; ++mt) {
#pragma unroll
    for (int nt = 0; nt < 4; ++nt) {
      int ncol = wave * 64 + nt * 16 + ln;
      float bb = b1[e * P_ + ncol];
#pragma unroll
      for (int r = 0; r < 8; ++r) {
        int m = mt * 16 + r + 8 * half;
        float v = acc[mt][nt][r] + bb;
        v = v > 0.f ? v : 0.f;
        hs[m * P_ + ncol] = f2bf(v * wts[m]);
      }
    }
  }
  __syncthreads();

  // ---------- GEMM2: out[32,D] += (w*h)[32,P] @ W2[P,D] + w*b2 ----------
  const unsigned short* W2e = W2t + (size_t)e * D_ * P_;
  const unsigned short* hrow0 = &hs[ln * P_];
  const unsigned short* hrow1 = &hs[(16 + ln) * P_];
  for (int j = 0; j < 16; ++j) {
    int ncol = (j * 8 + wave) * 16 + ln;   // column in D
    float bb2 = b2[e * D_ + ncol];
    v8f c0, c1;
#pragma unroll
    for (int r = 0; r < 8; ++r) {
      c0[r] = wts[r + 8 * half] * bb2;
      c1[r] = wts[16 + r + 8 * half] * bb2;
    }
#pragma unroll
    for (int kk = 0; kk < P_; kk += 32) {
      v16bf a0 = load_a(hrow0 + kk, half);
      v16bf a1 = load_a(hrow1 + kk, half);
      v16bf b  = load_b(W2e + (size_t)ncol * P_ + kk + half * 16);
      c0 = __builtin_amdgcn_wmma_f32_16x16x32_bf16(
          false, a0, false, b, (short)0, c0, false, false);
      c1 = __builtin_amdgcn_wmma_f32_16x16x32_bf16(
          false, a1, false, b, (short)0, c1, false, false);
    }
#pragma unroll
    for (int r = 0; r < 8; ++r) {
      int m0r = r + 8 * half;
      int m1r = 16 + r + 8 * half;
      int rg0 = rowids[m0r];
      int rg1 = rowids[m1r];
      if (rg0 >= 0) atomicAdd(&out[(size_t)rg0 * D_ + ncol], c0[r]);
      if (rg1 >= 0) atomicAdd(&out[(size_t)rg1 * D_ + ncol], c1[r]);
    }
  }
}

// ---------------- workspace layout (bytes) ----------------
#define OFF_ROWE   ((size_t)0)                       // 2B ints  : 65536
#define OFF_ROWW   ((size_t)65536)                   // 2B f32   : 65536
#define OFF_CNT    ((size_t)131072)                  // 8 ints
#define OFF_OFFS   ((size_t)131328)                  // 9 ints
#define OFF_CUR    ((size_t)131584)                  // 8 ints
#define OFF_PERM   ((size_t)131840)                  // 2B ints  : 65536
#define OFF_PERMW  ((size_t)197376)                  // 2B f32   : 65536
#define OFF_W1T    ((size_t)262912)                  // E*P*D bf16 : 16777216
#define OFF_W2T    ((size_t)(262912 + 16777216))     // E*D*P bf16 : 16777216

extern "C" void kernel_launch(void* const* d_in, const int* in_sizes, int n_in,
                              void* d_out, int out_size, void* d_ws, size_t ws_size,
                              hipStream_t stream) {
  const float* x   = (const float*)d_in[0];
  const float* Wg  = (const float*)d_in[1];
  const float* bg  = (const float*)d_in[2];
  const float* Wp  = (const float*)d_in[3];
  const float* bp  = (const float*)d_in[4];
  const float* Wgg = (const float*)d_in[5];
  const float* bgg = (const float*)d_in[6];
  const float* W1  = (const float*)d_in[7];
  const float* b1  = (const float*)d_in[8];
  const float* W2  = (const float*)d_in[9];
  const float* b2  = (const float*)d_in[10];
  float* out = (float*)d_out;

  char* ws = (char*)d_ws;
  int*   rowE   = (int*)(ws + OFF_ROWE);
  float* rowW   = (float*)(ws + OFF_ROWW);
  int*   counts = (int*)(ws + OFF_CNT);
  int*   offs   = (int*)(ws + OFF_OFFS);
  int*   cursor = (int*)(ws + OFF_CUR);
  int*   perm   = (int*)(ws + OFF_PERM);
  float* permW  = (float*)(ws + OFF_PERMW);
  unsigned short* W1t = (unsigned short*)(ws + OFF_W1T);
  unsigned short* W2t = (unsigned short*)(ws + OFF_W2T);

  // 0) zero output + counts
  init_kernel<<<4096, 256, 0, stream>>>(out, counts);

  // 1) gating: one wave per row
  gate_kernel<<<B_ / 8, 256, 0, stream>>>(x, Wg, bg, Wp, bp, Wgg, bgg,
                                          rowE, rowW, counts);

  // 2) scan + 3) scatter
  scan_kernel<<<1, 32, 0, stream>>>(counts, offs, cursor);
  scatter_kernel<<<B_ / 256, 256, 0, stream>>>(rowE, rowW, cursor, perm, permW);

  // 4) weight transpose+convert: W1 [E,D,P]->[E,P,D], W2 [E,P,D]->[E,D,P]
  {
    dim3 blk(32, 8, 1);
    dim3 g1(P_ / 32, D_ / 32, E_);
    transpose_bf16<<<g1, blk, 0, stream>>>(W1, W1t, D_, P_);
    dim3 g2(D_ / 32, P_ / 32, E_);
    transpose_bf16<<<g2, blk, 0, stream>>>(W2, W2t, P_, D_);
  }

  // 5) routed expert MLP (bf16 WMMA, 32-row tiles)
  moe_gemm_kernel<<<E_ * MAX_TILES, 256, 0, stream>>>(
      x, W1t, b1, W2t, b2, counts, offs, perm, permW, out);
}
